// DiffHead_58531814310245
// MI455X (gfx1250) — matrix operand order repaired and checked by reading, not compiled, not run on hardware
//
#include <hip/hip_runtime.h>

typedef __bf16 bf16;
typedef bf16  v16bf __attribute__((ext_vector_type(16)));
typedef float v8f   __attribute__((ext_vector_type(8)));
typedef unsigned v4u __attribute__((ext_vector_type(4)));
typedef unsigned v8u __attribute__((ext_vector_type(8)));

union V16 {
    v16bf v;
    uint4 q[2];
    bf16  e[16];
};

#define SCALE_ATT 0.08838834764831845f   // 128^-0.5
#define LAMBDA_INIT 0.8f

// ---------------------------------------------------------------------------
// Projection GEMM:  X[8192,1024] (f32) @ W[1024,N] (f32)  -> bf16 out
// ---------------------------------------------------------------------------
__global__ __launch_bounds__(256)
void proj_kernel(const float* __restrict__ q, const float* __restrict__ k,
                 const float* __restrict__ v,
                 const float* __restrict__ Wq, const float* __restrict__ Wk,
                 const float* __restrict__ Wv,
                 bf16* __restrict__ Qb, bf16* __restrict__ Kb, bf16* __restrict__ Vb)
{
    __shared__ bf16 Wt[16][1024];   // transposed weight tile (32 KB)

    const int tid  = threadIdx.x;
    const int lane = tid & 31;
    const int wave = tid >> 5;
    const int half = lane >> 4;
    const int l15  = lane & 15;

    const int nt     = blockIdx.x % 40;
    const int rchunk = blockIdx.x / 40;

    const float* X;  const float* W;  bf16* Out;  int Nw;  int n0;
    if (nt < 16)      { X = q; W = Wq; Out = Qb; Nw = 256; n0 = nt * 16; }
    else if (nt < 32) { X = k; W = Wk; Out = Kb; Nw = 256; n0 = (nt - 16) * 16; }
    else              { X = v; W = Wv; Out = Vb; Nw = 128; n0 = (nt - 32) * 16; }

    for (int idx = tid; idx < 1024 * 16; idx += 256) {
        const int kk = idx >> 4;
        const int n  = idx & 15;
        Wt[n][kk] = (bf16)W[(long)kk * Nw + n0 + n];
    }
    __syncthreads();

    for (int i = 0; i < 8; ++i) {
        const int rt = rchunk * 64 + wave * 8 + i;
        v8f acc = {0.f, 0.f, 0.f, 0.f, 0.f, 0.f, 0.f, 0.f};

        for (int kb = 0; kb < 1024; kb += 32) {
            const float* arow = X + ((long)(rt * 16 + l15) * 1024 + kb + half * 8);
            __builtin_prefetch(arow + 256, 0, 0);          // global_prefetch_b8
            const float4 f0 = ((const float4*)arow)[0];
            const float4 f1 = ((const float4*)arow)[1];
            const float4 f2 = ((const float4*)(arow + 16))[0];
            const float4 f3 = ((const float4*)(arow + 16))[1];
            V16 a;
            a.e[0]  = (bf16)f0.x; a.e[1]  = (bf16)f0.y; a.e[2]  = (bf16)f0.z; a.e[3]  = (bf16)f0.w;
            a.e[4]  = (bf16)f1.x; a.e[5]  = (bf16)f1.y; a.e[6]  = (bf16)f1.z; a.e[7]  = (bf16)f1.w;
            a.e[8]  = (bf16)f2.x; a.e[9]  = (bf16)f2.y; a.e[10] = (bf16)f2.z; a.e[11] = (bf16)f2.w;
            a.e[12] = (bf16)f3.x; a.e[13] = (bf16)f3.y; a.e[14] = (bf16)f3.z; a.e[15] = (bf16)f3.w;

            V16 b;
            const bf16* wp = &Wt[l15][kb + half * 8];
            b.q[0] = *(const uint4*)wp;
            b.q[1] = *(const uint4*)(wp + 16);

            acc = __builtin_amdgcn_wmma_f32_16x16x32_bf16(
                false, a.v, false, b.v, (short)0, acc, false, false);
        }

        #pragma unroll
        for (int r = 0; r < 8; ++r) {
            const long rowg = (long)rt * 16 + r + half * 8;
            Out[rowg * Nw + n0 + l15] = (bf16)acc[r];
        }
    }
}

// ---------------------------------------------------------------------------
// Flash differential attention with TDM K-staging + async V-staging.
// Grid: 64 blocks = 4 batches * 16 query-tiles(128). 8 waves * 16 queries.
// ---------------------------------------------------------------------------
__global__ __launch_bounds__(256)
void attn_kernel(const bf16* __restrict__ Qb, const bf16* __restrict__ Kb,
                 const bf16* __restrict__ Vb,
                 const float* __restrict__ lq1, const float* __restrict__ lk1,
                 const float* __restrict__ lq2, const float* __restrict__ lk2,
                 float* __restrict__ out)
{
    __shared__ bf16 KC[32][256];         // 16 KB: K1 = cols 0..127, K2 = 128..255
    __shared__ bf16 Vrow[32][128];       //  8 KB: V chunk, row-major
    __shared__ bf16 Pbuf[8][16][32];     //  8 KB: per-wave P staging
    __shared__ float s_lbd;

    const int tid  = threadIdx.x;
    const int lane = tid & 31;
    const int wave = tid >> 5;
    const int half = lane >> 4;
    const int l15  = lane & 15;

    const int b   = blockIdx.x >> 4;
    const int qt  = blockIdx.x & 15;
    const int qb0 = qt * 128;
    const int qw0 = qb0 + wave * 16;

    if (tid == 0) {
        float d1 = 0.f, d2 = 0.f;
        for (int i = 0; i < 128; ++i) {
            d1 += lq1[i] * lk1[i];
            d2 += lq2[i] * lk2[i];
        }
        s_lbd = __expf(d1) - __expf(d2) + LAMBDA_INIT;
    }

    // Preload Q1/Q2 A-fragments.
    V16 q1a[4], q2a[4];
    {
        const long rowQ = (long)b * 2048 + qw0 + l15;
        #pragma unroll
        for (int kt = 0; kt < 4; ++kt) {
            const bf16* base1 = Qb + rowQ * 256 + kt * 32 + half * 16;
            q1a[kt].q[0] = *(const uint4*)(base1);
            q1a[kt].q[1] = *(const uint4*)(base1 + 8);
            const bf16* base2 = base1 + 128;
            q2a[kt].q[0] = *(const uint4*)(base2);
            q2a[kt].q[1] = *(const uint4*)(base2 + 8);
        }
    }

    v8f O1[8], O2[8];
    float m1[8], l1[8], m2[8], l2[8];
    const v8f vzero = {0.f, 0.f, 0.f, 0.f, 0.f, 0.f, 0.f, 0.f};
    #pragma unroll
    for (int h = 0; h < 8; ++h) { O1[h] = vzero; O2[h] = vzero; }
    #pragma unroll
    for (int r = 0; r < 8; ++r) { m1[r] = -3e38f; m2[r] = -3e38f; l1[r] = 0.f; l2[r] = 0.f; }

    const int qmaxp1 = qw0 + 16;
    const int kmax   = (2048 < qb0 + 129) ? 2048 : qb0 + 129;
    const int nchunk = (kmax + 31) >> 5;

    __syncthreads();   // s_lbd visible

    auto attend = [&](const v8f& stA, const v8f& stB, float* mr, float* lr,
                      v8f* O, const V16* vb, int ck) {
        #pragma unroll
        for (int r = 0; r < 8; ++r) {
            const int lim = qw0 + r + half * 8 + 1;        // j <= i+1
            const int j0  = ck + l15;
            const int j1  = ck + 16 + l15;
            const bool ok0 = j0 <= lim, ok1 = j1 <= lim;
            const float a0 = stA[r] * SCALE_ATT;
            const float a1 = stB[r] * SCALE_ATT;
            float rmax = fmaxf(ok0 ? a0 : -3e38f, ok1 ? a1 : -3e38f);
            rmax = fmaxf(rmax, __shfl_xor(rmax, 1, 32));
            rmax = fmaxf(rmax, __shfl_xor(rmax, 2, 32));
            rmax = fmaxf(rmax, __shfl_xor(rmax, 4, 32));
            rmax = fmaxf(rmax, __shfl_xor(rmax, 8, 32));
            const float mnew = fmaxf(mr[r], rmax);
            const float sc   = __expf(mr[r] - mnew);
            const float p0   = ok0 ? __expf(a0 - mnew) : 0.f;
            const float p1   = ok1 ? __expf(a1 - mnew) : 0.f;
            float ps = p0 + p1;
            ps += __shfl_xor(ps, 1, 32);
            ps += __shfl_xor(ps, 2, 32);
            ps += __shfl_xor(ps, 4, 32);
            ps += __shfl_xor(ps, 8, 32);
            lr[r] = lr[r] * sc + ps;
            mr[r] = mnew;
            #pragma unroll
            for (int h = 0; h < 8; ++h) O[h][r] *= sc;
            Pbuf[wave][r + half * 8][l15]      = (bf16)p0;
            Pbuf[wave][r + half * 8][l15 + 16] = (bf16)p1;
        }
        V16 pa;
        const bf16* pb = &Pbuf[wave][l15][half * 8];
        pa.q[0] = *(const uint4*)pb;
        pa.q[1] = *(const uint4*)(pb + 16);
        #pragma unroll
        for (int h = 0; h < 8; ++h) {
            O[h] = __builtin_amdgcn_wmma_f32_16x16x32_bf16(
                false, pa.v, false, vb[h].v, (short)0, O[h], false, false);
        }
    };

    for (int c = 0; c < nchunk; ++c) {
        const int ck = c * 32;

        __syncthreads();   // previous chunk's LDS reads done

        // --- K chunk via Tensor Data Mover: contiguous 32x256 bf16 tile ---
        if (wave == 0) {
            const unsigned long long ga =
                (unsigned long long)(Kb + ((long)b * 2048 + ck) * 256);
            const unsigned navail = (unsigned)(8192 - (b * 2048 + ck)); // OOB rows zero-fill
            const unsigned kc_lds = (unsigned)(unsigned long long)&KC[0][0];
            v4u g0;
            g0[0] = 1u;                                        // count=1, user D#
            g0[1] = kc_lds;                                    // lds_addr
            g0[2] = (unsigned)(ga & 0xFFFFFFFFull);            // global_addr[31:0]
            g0[3] = (unsigned)((ga >> 32) & 0x01FFFFFFull)     // global_addr[56:32]
                    | (2u << 30);                              // type=2 (image)
            v8u g1;
            g1[0] = 1u << 16;                                  // data_size=1 -> 2 bytes
            g1[1] = (256u & 0xFFFFu) << 16;                    // tensor_dim0[15:0]
            g1[2] = (256u >> 16) | ((navail & 0xFFFFu) << 16); // dim0[31:16] | dim1[15:0]
            g1[3] = (navail >> 16) | (256u << 16);             // dim1[31:16] | tile_dim0
            g1[4] = 32u;                                       // tile_dim1=32, tile_dim2=0
            g1[5] = 256u;                                      // tensor_dim0_stride[31:0]
            g1[6] = 0u;                                        // stride0[47:32] | stride1[15:0]
            g1[7] = 0u;
            asm volatile("tensor_load_to_lds %0, %1"
                         :
                         : "s"(g0), "s"(g1)
                         : "memory");
        }

        // --- V chunk via async global->LDS copies (row-major) ---
        for (int idx = tid; idx < 512; idx += 256) {
            const int r = idx >> 4, u = idx & 15;
            int krow = ck + r; if (krow > 2047) krow = 2047;   // masked anyway
            const unsigned long long ga =
                (unsigned long long)(Vb + ((long)b * 2048 + krow) * 128 + u * 8);
            const unsigned ldsa = (unsigned)(unsigned long long)&Vrow[r][u * 8];
            asm volatile("global_load_async_to_lds_b128 %0, %1, off"
                         :
                         : "v"(ldsa), "v"(ga)
                         : "memory");
        }

        asm volatile("s_wait_asynccnt 0" ::: "memory");
        if (wave == 0) __builtin_amdgcn_s_wait_tensorcnt(0);
        __syncthreads();

        if (ck <= qmaxp1) {
            // V B-fragments (transpose at read time), shared by O1 and O2.
            V16 vb[8];
            #pragma unroll
            for (int h = 0; h < 8; ++h) {
                #pragma unroll
                for (int e = 0; e < 16; ++e)
                    vb[h].e[e] = Vrow[(e & 7) + 8 * half + 16 * (e >> 3)][h * 16 + l15];
            }

            v8f s1t0 = vzero, s1t1 = vzero, s2t0 = vzero, s2t1 = vzero;
            #pragma unroll
            for (int kt = 0; kt < 4; ++kt) {
                V16 b1, b2;
                const bf16* kp1 = &KC[l15][kt * 32 + half * 16];
                b1.q[0] = *(const uint4*)kp1;  b1.q[1] = *(const uint4*)(kp1 + 8);
                const bf16* kp2 = &KC[l15][128 + kt * 32 + half * 16];
                b2.q[0] = *(const uint4*)kp2;  b2.q[1] = *(const uint4*)(kp2 + 8);
                s1t0 = __builtin_amdgcn_wmma_f32_16x16x32_bf16(
                    false, q1a[kt].v, false, b1.v, (short)0, s1t0, false, false);
                s2t0 = __builtin_amdgcn_wmma_f32_16x16x32_bf16(
                    false, q2a[kt].v, false, b2.v, (short)0, s2t0, false, false);
                V16 c1, c2;
                const bf16* kq1 = &KC[16 + l15][kt * 32 + half * 16];
                c1.q[0] = *(const uint4*)kq1;  c1.q[1] = *(const uint4*)(kq1 + 8);
                const bf16* kq2 = &KC[16 + l15][128 + kt * 32 + half * 16];
                c2.q[0] = *(const uint4*)kq2;  c2.q[1] = *(const uint4*)(kq2 + 8);
                s1t1 = __builtin_amdgcn_wmma_f32_16x16x32_bf16(
                    false, q1a[kt].v, false, c1.v, (short)0, s1t1, false, false);
                s2t1 = __builtin_amdgcn_wmma_f32_16x16x32_bf16(
                    false, q2a[kt].v, false, c2.v, (short)0, s2t1, false, false);
            }
            attend(s1t0, s1t1, m1, l1, O1, vb, ck);
            attend(s2t0, s2t1, m2, l2, O2, vb, ck);
        }
    }

    const float lbd = s_lbd;
    #pragma unroll
    for (int h = 0; h < 8; ++h) {
        #pragma unroll
        for (int r = 0; r < 8; ++r) {
            const long row = (long)b * 2048 + qw0 + r + half * 8;
            const float o = O1[h][r] / l1[r] - lbd * (O2[h][r] / l2[r]);
            out[row * 128 + h * 16 + l15] = o;
        }
    }
}

// ---------------------------------------------------------------------------
extern "C" void kernel_launch(void* const* d_in, const int* in_sizes, int n_in,
                              void* d_out, int out_size, void* d_ws, size_t ws_size,
                              hipStream_t stream)
{
    const float* q   = (const float*)d_in[0];
    const float* k   = (const float*)d_in[1];
    const float* v   = (const float*)d_in[2];
    const float* Wq  = (const float*)d_in[3];
    const float* Wk  = (const float*)d_in[4];
    const float* Wv  = (const float*)d_in[5];
    const float* lq1 = (const float*)d_in[6];
    const float* lk1 = (const float*)d_in[7];
    const float* lq2 = (const float*)d_in[8];
    const float* lk2 = (const float*)d_in[9];

    bf16* Qb = (bf16*)d_ws;                         // [8192, 256]
    bf16* Kb = Qb + (size_t)8192 * 256;             // [8192, 256]
    bf16* Vb = Kb + (size_t)8192 * 256;             // [8192, 128]

    proj_kernel<<<320, 256, 0, stream>>>(q, k, v, Wq, Wk, Wv, Qb, Kb, Vb);
    attn_kernel<<<64, 256, 0, stream>>>(Qb, Kb, Vb, lq1, lk1, lq2, lk2,
                                        (float*)d_out);
}
